// LocalKConv_10960756539626
// MI455X (gfx1250) — compile-verified, matrix operand order checked by reading, not run on hardware
//
#include <hip/hip_runtime.h>
#include <hip/hip_bf16.h>
#include <math.h>

typedef __attribute__((ext_vector_type(16))) _Float16 v16h;
typedef __attribute__((ext_vector_type(8)))  _Float16 v8h;
typedef __attribute__((ext_vector_type(8)))  float    v8f;
typedef __attribute__((ext_vector_type(4)))  unsigned int u32x4;
typedef __attribute__((ext_vector_type(4)))  int      i32x4;
typedef __attribute__((ext_vector_type(8)))  int      i32x8;

#define BATCH 8
#define CCH   128
#define WW    56
#define HWP   3136      // 56*56
#define WPAD  58
#define CB    384       // 3*C basis channels
#define K3    3456      // 384*9
#define LNEPS 1e-5f

#if defined(__has_builtin)
#  if __has_builtin(__builtin_amdgcn_tensor_load_to_lds) && \
      __has_builtin(__builtin_amdgcn_s_wait_tensorcnt)
#    define KAN_HAVE_TDM 1
#  endif
#endif
#ifndef KAN_HAVE_TDM
#  define KAN_HAVE_TDM 0
#endif

__device__ __forceinline__ v8f wmma16(v16h a, v16h b, v8f c) {
  return __builtin_amdgcn_wmma_f32_16x16x32_f16(false, a, false, b, (short)0, c,
                                                false, false);
}

// Build a 16-half WMMA fragment from LDS: lane = 16*k_half + row/col,
// halfs [k_half*8 .. +7] and [16 + k_half*8 .. +7] (two ds_load_b128).
__device__ __forceinline__ v16h frag_from_lds(const _Float16* p) {
  v8h lo = *(const v8h*)p;
  v8h hi = *(const v8h*)(p + 16);
  v16h f;
#pragma unroll
  for (int i = 0; i < 8; ++i) { f[i] = lo[i]; f[8 + i] = hi[i]; }
  return f;
}

#if KAN_HAVE_TDM
// Issue a 2D TDM load: rows x 32 fp16 tile from a row-major [rows][ldK]
// fp16 matrix (tile column k0) into LDS at ldsOff. One issue per wave.
__device__ __forceinline__ void tdm_load_tile(const _Float16* gsrc, unsigned ldsOff,
                                              int ldK, int rows) {
  unsigned long long ga = (unsigned long long)(unsigned long long)gsrc;
  u32x4 g0;
  g0[0] = 1u;                                   // count=1, user mode
  g0[1] = ldsOff;                               // lds_addr
  g0[2] = (unsigned)ga;                         // global_addr[31:0]
  g0[3] = (unsigned)((ga >> 32) & 0x1FFFFFFu) | (2u << 30);  // addr[56:32]|type=2
  i32x8 g1;
  g1[0] = 0x10000;                              // data_size=1 (2 bytes)
  g1[1] = (int)((unsigned)(ldK & 0xFFFF) << 16);        // tensor_dim0 lo
  g1[2] = (int)((ldK >> 16) | ((rows & 0xFFFF) << 16)); // dim0 hi | tensor_dim1 lo
  g1[3] = (int)(32u << 16);                     // tensor_dim1 hi=0 | tile_dim0=32
  g1[4] = rows;                                 // tile_dim1 | tile_dim2=0
  g1[5] = ldK;                                  // tensor_dim0_stride lo32
  g1[6] = 0;                                    // stride hi | dim1_stride lo
  g1[7] = 0;
  i32x4 gz = {0, 0, 0, 0};
#if __clang_major__ >= 23
  i32x8 gz8 = {0, 0, 0, 0, 0, 0, 0, 0};
  __builtin_amdgcn_tensor_load_to_lds(g0, g1, gz, gz, gz8, 0);
#else
  __builtin_amdgcn_tensor_load_to_lds(g0, g1, gz, gz, 0);
#endif
}
#endif

// ---------------------------------------------------------------- utilities
__global__ void __launch_bounds__(256) zero_u32_kernel(unsigned int* p, long n) {
  long i = (long)blockIdx.x * blockDim.x + threadIdx.x;
  if (i < n) p[i] = 0u;
}

// kan_w [128][384][3][3] fp32 -> A [co][k] fp16, k = (kh*3+kw)*384 + ci
__global__ void __launch_bounds__(256) pack_w3_kernel(const float* __restrict__ w,
                                                      _Float16* __restrict__ a) {
  long i = (long)blockIdx.x * blockDim.x + threadIdx.x;
  if (i >= (long)CCH * K3) return;
  int k  = (int)(i % K3);
  int co = (int)(i / K3);
  int tap = k / CB, ci = k % CB;
  int kh = tap / 3, kw = tap % 3;
  a[i] = (_Float16)w[(((long)co * CB + ci) * 3 + kh) * 3 + kw];
}

__global__ void __launch_bounds__(256) pack_w2_kernel(const float* __restrict__ w,
                                                      _Float16* __restrict__ a) {
  int i = blockIdx.x * blockDim.x + threadIdx.x;
  if (i < CCH * CCH) a[i] = (_Float16)w[i];
}

// ------------------------------------------- LN1 + tanh + basis (padded fp16)
__global__ void __launch_bounds__(256) ln_basis_kernel(const float* __restrict__ x,
                                                       const float* __restrict__ g,
                                                       const float* __restrict__ be,
                                                       _Float16* __restrict__ basis) {
  int idx = blockIdx.x * blockDim.x + threadIdx.x;
  if (idx >= BATCH * HWP) return;
  int b = idx / HWP, p = idx % HWP;
  int y = p / WW, xx = p % WW;
  const float* xp = x + (long)b * CCH * HWP + p;
  float s = 0.f, s2 = 0.f;
  for (int c = 0; c < CCH; ++c) { float v = xp[(long)c * HWP]; s += v; s2 += v * v; }
  float mean = s * (1.f / CCH);
  float var  = s2 * (1.f / CCH) - mean * mean;
  float rden = 1.f / (sqrtf(var) + LNEPS);
  _Float16* bp = basis + (long)b * CB * WPAD * WPAD + (long)(y + 1) * WPAD + (xx + 1);
  const long cs = (long)WPAD * WPAD;
  for (int c = 0; c < CCH; ++c) {
    float t  = (xp[(long)c * HWP] - mean) * rden * g[c] + be[c];
    float th = tanhf(t);
    bp[(long)c * cs]              = (_Float16)1.0f;
    bp[(long)(CCH + c) * cs]      = (_Float16)th;
    bp[(long)(2 * CCH + c) * cs]  = (_Float16)(2.f * th * th - 1.f);
  }
}

// ------------------------------------------------ GEMM1: 3x3 conv via WMMA
// Block: 128(M) x 64(N), 8 waves, each wave a 32x32 tile (4 WMMA / K-step).
__global__ void __launch_bounds__(256) kan_gemm3x3_kernel(
    const _Float16* __restrict__ A,    // [128][3456]
    const _Float16* __restrict__ Bp,   // [8][384][58][58] zero-padded basis
    _Float16* __restrict__ Mid)        // [8][128][3136] fp16
{
  __shared__ _Float16 sA[128 * 32];    // [m][k]  8KB
  __shared__ _Float16 sB[64 * 32];     // [n][k]  4KB (transposed on store)
  const int b  = blockIdx.z;
  const int n0 = blockIdx.x * 64;
  const int tid  = threadIdx.x;
  const int wave = tid >> 5, lane = tid & 31;
  const int wm = (wave & 3) * 32;
  const int wn = (wave >> 2) * 32;
  const int half = lane >> 4, l16 = lane & 15;

  v8f acc00 = {}, acc01 = {}, acc10 = {}, acc11 = {};
  const _Float16* bBase = Bp + (long)b * CB * WPAD * WPAD;
#if KAN_HAVE_TDM
  const unsigned ldsA = (unsigned)(unsigned long long)(void*)&sA[0];
#endif

  for (int k0 = 0; k0 < K3; k0 += 32) {
#if KAN_HAVE_TDM
    if (wave == 0) tdm_load_tile(A + k0, ldsA, K3, 128);   // async DMA -> sA
#else
    { // A tile 128x32, 16 halfs per thread
      int r = tid >> 1, c = (tid & 1) * 16;
      const _Float16* src = A + (long)r * K3 + k0 + c;
      __builtin_prefetch(src + 32, 0, 1);
      *(v8h*)&sA[r * 32 + c]     = *(const v8h*)src;
      *(v8h*)&sA[r * 32 + c + 8] = *(const v8h*)(src + 8);
    }
#endif
    { // B tile 32x64 from padded basis; 8-aligned pixel runs never cross an
      // image row (8 | 56) -> one contiguous 16B (2B-aligned) load per thread
      int e0 = tid * 8;
      int kk = e0 >> 6, nn = e0 & 63;
      int tap = k0 / CB, ci0 = k0 % CB;
      int dy = tap / 3, dx = tap % 3;
      int n = n0 + nn;
      int y = n / WW, xx = n % WW;
      const _Float16* src =
          bBase + ((long)(ci0 + kk) * WPAD + (y + dy)) * WPAD + (xx + dx);
      __builtin_prefetch(src + 32 * WPAD * WPAD, 0, 1);
      v8h v;
      __builtin_memcpy(&v, src, 16);
#pragma unroll
      for (int i = 0; i < 8; ++i) sB[(nn + i) * 32 + kk] = v[i];
    }
#if KAN_HAVE_TDM
    if (wave == 0) __builtin_amdgcn_s_wait_tensorcnt(0);
#endif
    __syncthreads();
    v16h af0 = frag_from_lds(&sA[(wm + l16) * 32 + half * 8]);
    v16h af1 = frag_from_lds(&sA[(wm + 16 + l16) * 32 + half * 8]);
    v16h bf0 = frag_from_lds(&sB[(wn + l16) * 32 + half * 8]);
    v16h bf1 = frag_from_lds(&sB[(wn + 16 + l16) * 32 + half * 8]);
    acc00 = wmma16(af0, bf0, acc00);
    acc01 = wmma16(af0, bf1, acc01);
    acc10 = wmma16(af1, bf0, acc10);
    acc11 = wmma16(af1, bf1, acc11);
    __syncthreads();
  }
  // C/D layout: VGPR r -> M = half*8 + r, N = l16
  _Float16* out = Mid + (long)b * CCH * HWP;
#pragma unroll
  for (int r = 0; r < 8; ++r) {
    int r0 = wm + half * 8 + r, r1 = r0 + 16;
    int c0 = n0 + wn + l16;
    out[(long)r0 * HWP + c0]      = (_Float16)acc00[r];
    out[(long)r0 * HWP + c0 + 16] = (_Float16)acc01[r];
    out[(long)r1 * HWP + c0]      = (_Float16)acc10[r];
    out[(long)r1 * HWP + c0 + 16] = (_Float16)acc11[r];
  }
}

// ----------------------------- GEMM2: 1x1 conv + bias + residual (WMMA, K=128)
__global__ void __launch_bounds__(256) kan_gemm1x1_kernel(
    const _Float16* __restrict__ A,    // [128][128]
    const _Float16* __restrict__ Mid,  // [8][128][3136]
    const float* __restrict__ x,
    const float* __restrict__ bias,
    float* __restrict__ Pre)           // [8][128][3136] fp32 pre-LN2
{
  __shared__ _Float16 sA[128 * 32];
  __shared__ _Float16 sB[64 * 32];
  const int b  = blockIdx.z;
  const int n0 = blockIdx.x * 64;
  const int tid  = threadIdx.x;
  const int wave = tid >> 5, lane = tid & 31;
  const int wm = (wave & 3) * 32;
  const int wn = (wave >> 2) * 32;
  const int half = lane >> 4, l16 = lane & 15;

  v8f acc00 = {}, acc01 = {}, acc10 = {}, acc11 = {};
  const _Float16* mB = Mid + (long)b * CCH * HWP;

  for (int k0 = 0; k0 < CCH; k0 += 32) {
    {
      int r = tid >> 1, c = (tid & 1) * 16;
      const _Float16* src = A + (long)r * CCH + k0 + c;
      *(v8h*)&sA[r * 32 + c]     = *(const v8h*)src;
      *(v8h*)&sA[r * 32 + c + 8] = *(const v8h*)(src + 8);
    }
    {
      int e0 = tid * 8;
      int kk = e0 >> 6, nn = e0 & 63;
      v8h v = *(const v8h*)(mB + (long)(k0 + kk) * HWP + n0 + nn);
#pragma unroll
      for (int i = 0; i < 8; ++i) sB[(nn + i) * 32 + kk] = v[i];
    }
    __syncthreads();
    v16h af0 = frag_from_lds(&sA[(wm + l16) * 32 + half * 8]);
    v16h af1 = frag_from_lds(&sA[(wm + 16 + l16) * 32 + half * 8]);
    v16h bf0 = frag_from_lds(&sB[(wn + l16) * 32 + half * 8]);
    v16h bf1 = frag_from_lds(&sB[(wn + 16 + l16) * 32 + half * 8]);
    acc00 = wmma16(af0, bf0, acc00);
    acc01 = wmma16(af0, bf1, acc01);
    acc10 = wmma16(af1, bf0, acc10);
    acc11 = wmma16(af1, bf1, acc11);
    __syncthreads();
  }
  float* out = Pre + (long)b * CCH * HWP;
  const float* xb = x + (long)b * CCH * HWP;
#pragma unroll
  for (int r = 0; r < 8; ++r) {
    int r0 = wm + half * 8 + r, r1 = r0 + 16;
    int c0 = n0 + wn + l16;
    long o00 = (long)r0 * HWP + c0, o10 = (long)r1 * HWP + c0;
    out[o00]      = acc00[r] + bias[r0] + xb[o00];
    out[o00 + 16] = acc01[r] + bias[r0] + xb[o00 + 16];
    out[o10]      = acc10[r] + bias[r1] + xb[o10];
    out[o10 + 16] = acc11[r] + bias[r1] + xb[o10 + 16];
  }
}

// -------------------------------------------- LN2 + exact GELU + residual
__global__ void __launch_bounds__(256) ln_gelu_kernel(const float* __restrict__ pre,
                                                      const float* __restrict__ x,
                                                      const float* __restrict__ g,
                                                      const float* __restrict__ be,
                                                      float* __restrict__ out) {
  int idx = blockIdx.x * blockDim.x + threadIdx.x;
  if (idx >= BATCH * HWP) return;
  int b = idx / HWP, p = idx % HWP;
  const float* pp = pre + (long)b * CCH * HWP + p;
  const float* xp = x   + (long)b * CCH * HWP + p;
  float*       op = out + (long)b * CCH * HWP + p;
  float s = 0.f, s2 = 0.f;
  for (int c = 0; c < CCH; ++c) { float v = pp[(long)c * HWP]; s += v; s2 += v * v; }
  float mean = s * (1.f / CCH);
  float var  = s2 * (1.f / CCH) - mean * mean;
  float rden = 1.f / (sqrtf(var) + LNEPS);
  for (int c = 0; c < CCH; ++c) {
    float t  = (pp[(long)c * HWP] - mean) * rden * g[c] + be[c];
    float ge = t * 0.5f * (1.f + erff(t * 0.70710678118654752f));
    op[(long)c * HWP] = ge + xp[(long)c * HWP];
  }
}

// ---------------------------------------------------------------- launcher
extern "C" void kernel_launch(void* const* d_in, const int* in_sizes, int n_in,
                              void* d_out, int out_size, void* d_ws, size_t ws_size,
                              hipStream_t stream) {
  const float* x    = (const float*)d_in[0];
  const float* g    = (const float*)d_in[1];
  const float* be   = (const float*)d_in[2];
  const float* kanw = (const float*)d_in[3];
  const float* w2   = (const float*)d_in[4];
  const float* b2   = (const float*)d_in[5];
  float* out = (float*)d_out;

  char* ws = (char*)d_ws;
  const long nBasis = (long)BATCH * CB * WPAD * WPAD;       // fp16 elems
  size_t off = 0;
  _Float16* basis = (_Float16*)(ws + off);
  off += ((size_t)nBasis * 2 + 255) & ~(size_t)255;
  _Float16* wa = (_Float16*)(ws + off);
  off += (((size_t)CCH * K3) * 2 + 255) & ~(size_t)255;
  _Float16* w2h = (_Float16*)(ws + off);
  off += (((size_t)CCH * CCH) * 2 + 255) & ~(size_t)255;
  _Float16* mid = (_Float16*)(ws + off);
  off += (((size_t)BATCH * CCH * HWP) * 2 + 255) & ~(size_t)255;
  float* pre = (float*)(ws + off);

  long nz = nBasis / 2; // u32 count
  zero_u32_kernel<<<(unsigned)((nz + 255) / 256), 256, 0, stream>>>(
      (unsigned int*)basis, nz);
  long nw3 = (long)CCH * K3;
  pack_w3_kernel<<<(unsigned)((nw3 + 255) / 256), 256, 0, stream>>>(kanw, wa);
  pack_w2_kernel<<<(CCH * CCH + 255) / 256, 256, 0, stream>>>(w2, w2h);

  int npx = BATCH * HWP;
  ln_basis_kernel<<<(npx + 255) / 256, 256, 0, stream>>>(x, g, be, basis);

  dim3 grid(HWP / 64, 1, BATCH);   // 49 x 1 x 8, each block = 128M x 64N
  kan_gemm3x3_kernel<<<grid, 256, 0, stream>>>(wa, basis, mid);
  kan_gemm1x1_kernel<<<grid, 256, 0, stream>>>(w2h, mid, x, b2, pre);

  ln_gelu_kernel<<<(npx + 255) / 256, 256, 0, stream>>>(pre, x, g, be, out);
}
